// genesys_quantized_linear_37280316129794
// MI455X (gfx1250) — matrix operand order repaired and checked by reading, not compiled, not run on hardware
//
#include <hip/hip_runtime.h>
#include <hip/hip_bf16.h>
#include <math.h>

// ---------------------------------------------------------------------------
// Quantized-linear GEMM for MI455X (gfx1250, wave32):
//   out = (x @ W + b) * scale     [M=N=K=4096, fp32 in/out]
//
// bf16x3 split-precision WMMA (v_wmma_f32_16x16x32_bf16):
//   x = xh + xl, W = wh + wl  (bf16 RNE split)  ->  acc += xh*wh + xh*wl + xl*wh
//
// Preferred path (ws_size >= 128 MiB):
//   pass 1: split x -> Xh/Xl [M][K] bf16; split+transpose W -> Wth/Wtl [N][K]
//   pass 2: GEMM; tiles staged LDS-side by the Tensor Data Mover
//           (tensor_load_to_lds, one descriptor per 128x32 tile, pad_enable
//            produces the 80B padded LDS rows), waited with s_wait_tensorcnt.
// Fallback (small ws): fused kernel converting fp32->bf16 hi/lo in-loop.
// ---------------------------------------------------------------------------

typedef __attribute__((ext_vector_type(16))) __bf16 v16bf;
typedef __attribute__((ext_vector_type(8)))  float  v8f;

#define BM 128
#define BN 128
#define BK 32
#define KPAD 40 // halfwords per LDS row: 80B, 16B-aligned, spreads banks

#if __has_builtin(__builtin_amdgcn_tensor_load_to_lds) && \
    __has_builtin(__builtin_amdgcn_s_wait_tensorcnt)
#define TDM_OK 1
#else
#define TDM_OK 0
#endif

union FragBF {
  uint4 q[2];
  v16bf v;
};
union F4 {
  float4 v;
  float  f[4];
};

__device__ __forceinline__ unsigned short bfbits(__bf16 b) {
  union { __bf16 b; unsigned short u; } c;
  c.b = b;
  return c.u;
}

// split two floats into packed bf16 hi / bf16 lo (RNE, hardware cvt path)
__device__ __forceinline__ void split2(float f0, float f1,
                                       unsigned int& h, unsigned int& l) {
  __bf16 h0 = (__bf16)f0, h1 = (__bf16)f1;
  float r0 = f0 - (float)h0, r1 = f1 - (float)h1;
  __bf16 l0 = (__bf16)r0, l1 = (__bf16)r1;
  h = (unsigned)bfbits(h0) | ((unsigned)bfbits(h1) << 16);
  l = (unsigned)bfbits(l0) | ((unsigned)bfbits(l1) << 16);
}

// ---------------------------------------------------------------------------
// Shared WMMA tile-compute: one BK=32 step for one wave (4x2 16x16 frags).
// A LDS rows [m][k] (hi/lo), B LDS rows [n][k] (hi/lo), both KPAD-strided.
// ---------------------------------------------------------------------------
__device__ __forceinline__ void wmma_step(
    const unsigned short (*Ah)[KPAD], const unsigned short (*Al)[KPAD],
    const unsigned short (*Bh)[KPAD], const unsigned short (*Bl)[KPAD],
    int wm, int wn, int lh, int ln, v8f (&acc)[4][2])
{
  FragBF bh[2], bl[2];
#pragma unroll
  for (int fn = 0; fn < 2; ++fn) {
    const int n = wn * 32 + fn * 16 + ln;
    const uint4* ph = (const uint4*)&Bh[n][0];
    const uint4* pl = (const uint4*)&Bl[n][0];
    bh[fn].q[0] = ph[lh];
    bh[fn].q[1] = ph[2 + lh];
    bl[fn].q[0] = pl[lh];
    bl[fn].q[1] = pl[2 + lh];
  }
#pragma unroll
  for (int fm = 0; fm < 4; ++fm) {
    // A frag: lane half h reads K chunks {8h..8h+7} and {16+8h..23+8h}
    const int m = wm * 64 + fm * 16 + ln;
    FragBF ah, al;
    const uint4* ph = (const uint4*)&Ah[m][0];
    const uint4* pl = (const uint4*)&Al[m][0];
    ah.q[0] = ph[lh];
    ah.q[1] = ph[2 + lh];
    al.q[0] = pl[lh];
    al.q[1] = pl[2 + lh];
#pragma unroll
    for (int fn = 0; fn < 2; ++fn) {
      acc[fm][fn] = __builtin_amdgcn_wmma_f32_16x16x32_bf16(
          false, ah.v, false, bh[fn].v, (short)0, acc[fm][fn], false, false);
      acc[fm][fn] = __builtin_amdgcn_wmma_f32_16x16x32_bf16(
          false, ah.v, false, bl[fn].v, (short)0, acc[fm][fn], false, false);
      acc[fm][fn] = __builtin_amdgcn_wmma_f32_16x16x32_bf16(
          false, al.v, false, bh[fn].v, (short)0, acc[fm][fn], false, false);
    }
  }
}

__device__ __forceinline__ void epilogue(
    const float* __restrict__ Bias, float* __restrict__ Out, int Ndim,
    int bm0, int bn0, int wm, int wn, int lh, int ln, float scale,
    v8f (&acc)[4][2])
{
#pragma unroll
  for (int fn = 0; fn < 2; ++fn) {
    const int gn = bn0 + wn * 32 + fn * 16 + ln;
    const float bias = Bias[gn];
#pragma unroll
    for (int fm = 0; fm < 4; ++fm) {
#pragma unroll
      for (int r = 0; r < 8; ++r) {
        const int gm = bm0 + wm * 64 + fm * 16 + lh * 8 + r;
        Out[(size_t)gm * Ndim + gn] = (acc[fm][fn][r] + bias) * scale;
      }
    }
  }
}

// ---------------------------------------------------------------------------
// TDM: one descriptor loads a 128x32 bf16 tile (rows of 64B) into LDS with
// 16B padding after every 64B -> 80B LDS row stride (KPAD=40 halfwords).
// D# per CDNA5 ISA 8.3/8.4 (group0: count/lds/global/type, group1: dims).
// This toolchain exposes the 6-arg builtin:
//   (u32x4 g0, i32x8 g1, i32x4 g2, i32x4 g3, i32x8 extra, i32 cpol)
// groups 2/3 zero (2D tensor, iterate off), extra group zero-filled.
// ---------------------------------------------------------------------------
#if TDM_OK
typedef __attribute__((ext_vector_type(4))) unsigned int u32x4;
typedef __attribute__((ext_vector_type(8))) int i32x8;
typedef __attribute__((ext_vector_type(4))) int i32x4;

__device__ __forceinline__ void tdm_load_tile(const unsigned short* gsrc,
                                              unsigned short* ldst,
                                              int kElems, int rows) {
  const unsigned lds = (unsigned)(unsigned long long)
      (__attribute__((address_space(3))) unsigned short*)ldst;
  const unsigned long long ga = (unsigned long long)gsrc;

  u32x4 g0 = {
      1u,                                               // count=1, user mode
      lds,                                              // lds_addr (bytes)
      (unsigned)ga,                                     // global_addr[31:0]
      ((unsigned)(ga >> 32) & 0x1FFFFFFu) | (2u << 30)  // ga[56:32] | type=2
  };
  i32x8 g1 = {
      // data_size=2B | pad_enable | pad_interval=16DW(64B) | pad_amount=4DW(16B)
      (int)((1u << 16) | (1u << 20) | (3u << 22) | (3u << 25)),
      (int)(((unsigned)kElems & 0xFFFFu) << 16),          // tensor_dim0 [15:0]
      (int)(((unsigned)kElems >> 16) |
            (((unsigned)rows & 0xFFFFu) << 16)),          // dim0 hi | dim1 lo
      (int)(((unsigned)rows >> 16) | ((unsigned)BK << 16)),// dim1 hi | tile_dim0
      (int)BM,                                            // tile_dim1=128
      (int)kElems,                                        // dim0_stride [31:0]
      0,                                                  // stride hi | dim1_stride
      0
  };
  i32x4 z4 = (i32x4)0;
  i32x8 z8 = (i32x8)0;
  __builtin_amdgcn_tensor_load_to_lds(g0, g1, z4, z4, z8, 0);
}
#endif

// ---------------------------------------------------------------------------
// Pre-pass 1: elementwise bf16 hi/lo split (for x, layout preserved)
// ---------------------------------------------------------------------------
__global__ __launch_bounds__(256)
void convert_split_kernel(const float* __restrict__ src,
                          unsigned short* __restrict__ hi,
                          unsigned short* __restrict__ lo)
{
  const size_t i = ((size_t)blockIdx.x * 256 + threadIdx.x) * 2;
  const float2 v = *(const float2*)(src + i);
  unsigned h, l;
  split2(v.x, v.y, h, l);
  *(unsigned*)(hi + i) = h;
  *(unsigned*)(lo + i) = l;
}

// ---------------------------------------------------------------------------
// Pre-pass 2: W [K][N] fp32 -> Wth/Wtl [N][K] bf16 (LDS-tiled transpose)
// ---------------------------------------------------------------------------
__global__ __launch_bounds__(256)
void convert_wT_kernel(const float* __restrict__ W,
                       unsigned short* __restrict__ Wth,
                       unsigned short* __restrict__ Wtl, int K, int N)
{
  __shared__ float tile[32][33];
  const int n0 = blockIdx.x * 32, k0 = blockIdx.y * 32;
  const int tx = threadIdx.x, ty = threadIdx.y; // 32 x 8
#pragma unroll
  for (int i = 0; i < 4; ++i) {
    const int k = ty + i * 8;
    tile[k][tx] = W[(size_t)(k0 + k) * N + n0 + tx];
  }
  __syncthreads();
#pragma unroll
  for (int i = 0; i < 4; ++i) {
    const int n = ty + i * 8;
    const float f = tile[tx][n]; // = W[k0+tx][n0+n]
    const __bf16 h = (__bf16)f;
    const float r = f - (float)h;
    const __bf16 l = (__bf16)r;
    Wth[(size_t)(n0 + n) * K + k0 + tx] = bfbits(h);
    Wtl[(size_t)(n0 + n) * K + k0 + tx] = bfbits(l);
  }
}

// ---------------------------------------------------------------------------
// Main GEMM, pre-converted operands, TDM-staged tiles
// ---------------------------------------------------------------------------
__global__ __launch_bounds__(256)
void qlinear_gemm_tdm(const unsigned short* __restrict__ Xh,
                      const unsigned short* __restrict__ Xl,
                      const unsigned short* __restrict__ Wth,
                      const unsigned short* __restrict__ Wtl,
                      const float* __restrict__ Bias,
                      float* __restrict__ Out,
                      int Mdim, int Ndim, int Kdim, float scale)
{
  __shared__ unsigned short Ah[2][BM][KPAD];
  __shared__ unsigned short Al[2][BM][KPAD];
  __shared__ unsigned short Bh[2][BN][KPAD];
  __shared__ unsigned short Bl[2][BN][KPAD];

  const int tid = threadIdx.x;
  const int bn0 = blockIdx.x * BN;
  const int bm0 = blockIdx.y * BM;

  const int lane = tid & 31;
  const int wave = tid >> 5;
  const int wm   = wave & 1;
  const int wn   = wave >> 1;
  const int lh   = lane >> 4;
  const int ln   = lane & 15;

  const int nk = Kdim / BK;
  v8f acc[4][2] = {};

  auto fill = [&](int p, int k0) {
#if TDM_OK
    if (wave == 0) {
      tdm_load_tile(Xh  + (size_t)bm0 * Kdim + k0, &Ah[p][0][0], Kdim, Mdim);
      tdm_load_tile(Xl  + (size_t)bm0 * Kdim + k0, &Al[p][0][0], Kdim, Mdim);
      tdm_load_tile(Wth + (size_t)bn0 * Kdim + k0, &Bh[p][0][0], Kdim, Ndim);
      tdm_load_tile(Wtl + (size_t)bn0 * Kdim + k0, &Bl[p][0][0], Kdim, Ndim);
    }
#else
#pragma unroll
    for (int i = 0; i < 2; ++i) {
      const int c = tid + 256 * i;      // 512 16B chunks per tile
      const int row = c >> 2, kc = c & 3;
      const size_t ga = (size_t)(bm0 + row) * Kdim + k0 + kc * 8;
      const size_t gb = (size_t)(bn0 + row) * Kdim + k0 + kc * 8;
      *(uint4*)&Ah[p][row][kc * 8] = *(const uint4*)(Xh + ga);
      *(uint4*)&Al[p][row][kc * 8] = *(const uint4*)(Xl + ga);
      *(uint4*)&Bh[p][row][kc * 8] = *(const uint4*)(Wth + gb);
      *(uint4*)&Bl[p][row][kc * 8] = *(const uint4*)(Wtl + gb);
    }
#endif
  };
  auto fillWait = [&]() {
#if TDM_OK
    if (wave == 0) __builtin_amdgcn_s_wait_tensorcnt(0);
#endif
  };

  fill(0, 0);
  fillWait();
  __syncthreads();

#pragma unroll 1
  for (int s = 0; s < nk; ++s) {
    const int p = s & 1;
    if (s + 1 < nk) fill(p ^ 1, (s + 1) * BK); // DMA overlaps compute
    wmma_step(Ah[p], Al[p], Bh[p], Bl[p], wm, wn, lh, ln, acc);
    if (s + 1 < nk) fillWait();
    __syncthreads();
  }

  epilogue(Bias, Out, Ndim, bm0, bn0, wm, wn, lh, ln, scale, acc);
}

// ---------------------------------------------------------------------------
// Fallback: fused conversion GEMM (when workspace is too small)
// ---------------------------------------------------------------------------
__global__ __launch_bounds__(256)
void qlinear_gemm_fused(const float* __restrict__ X,
                        const float* __restrict__ W,
                        const float* __restrict__ Bias,
                        float* __restrict__ Out,
                        int Mdim, int Ndim, int Kdim, float scale)
{
  __shared__ unsigned short Ah[2][BM][KPAD];
  __shared__ unsigned short Al[2][BM][KPAD];
  __shared__ unsigned short Bh[2][BN][KPAD];
  __shared__ unsigned short Bl[2][BN][KPAD];

  const int tid = threadIdx.x;
  const int bn0 = blockIdx.x * BN;
  const int bm0 = blockIdx.y * BM;

  const int ar  = tid >> 1;
  const int ac0 = (tid & 1) * 4;
  const int bnn = tid & 127;
  const int bq0 = (tid >> 7) * 4;

  const int lane = tid & 31;
  const int wave = tid >> 5;
  const int wm   = wave & 1;
  const int wn   = wave >> 1;
  const int lh   = lane >> 4;
  const int ln   = lane & 15;

  F4    aReg[4];
  float bReg[4][4];
  const int nk = Kdim / BK;

  auto loadTiles = [&](int k0) {
    const float4* pa = (const float4*)(X + (size_t)(bm0 + ar) * Kdim + k0);
#pragma unroll
    for (int i = 0; i < 4; ++i) aReg[i].v = pa[ac0 + i];
#pragma unroll
    for (int j = 0; j < 4; ++j) {
      const int k = (bq0 + j) * 4;
#pragma unroll
      for (int i = 0; i < 4; ++i)
        bReg[j][i] = W[(size_t)(k0 + k + i) * Ndim + bn0 + bnn];
    }
  };
  auto storeTiles = [&](int p) {
#pragma unroll
    for (int i = 0; i < 4; ++i) {
      const int k = (ac0 + i) * 4;
      uint2 h, l;
      split2(aReg[i].f[0], aReg[i].f[1], h.x, l.x);
      split2(aReg[i].f[2], aReg[i].f[3], h.y, l.y);
      *(uint2*)&Ah[p][ar][k] = h;
      *(uint2*)&Al[p][ar][k] = l;
    }
#pragma unroll
    for (int j = 0; j < 4; ++j) {
      const int k = (bq0 + j) * 4;
      uint2 h, l;
      split2(bReg[j][0], bReg[j][1], h.x, l.x);
      split2(bReg[j][2], bReg[j][3], h.y, l.y);
      *(uint2*)&Bh[p][bnn][k] = h;
      *(uint2*)&Bl[p][bnn][k] = l;
    }
  };

  v8f acc[4][2] = {};

  loadTiles(0);
  storeTiles(0);
  __syncthreads();

#pragma unroll 1
  for (int s = 0; s < nk; ++s) {
    const int p = s & 1;
    if (s + 1 < nk) loadTiles((s + 1) * BK);
    wmma_step(Ah[p], Al[p], Bh[p], Bl[p], wm, wn, lh, ln, acc);
    if (s + 1 < nk) storeTiles(p ^ 1);
    __syncthreads();
  }

  epilogue(Bias, Out, Ndim, bm0, bn0, wm, wn, lh, ln, scale, acc);
}

// ---------------------------------------------------------------------------
extern "C" void kernel_launch(void* const* d_in, const int* in_sizes, int n_in,
                              void* d_out, int out_size, void* d_ws, size_t ws_size,
                              hipStream_t stream) {
  const float* x = (const float*)d_in[0];
  const float* W = (const float*)d_in[1];
  const float* b = (const float*)d_in[2];
  float* out = (float*)d_out;

  const long long N = in_sizes[2];
  const long long K = (long long)in_sizes[1] / N;
  const long long M = (long long)in_sizes[0] / K;

  // TFLite-style requantization constant (host-side, static)
  double m = 0.0072;
  int rs = 0;
  while (m < 0.5) { m *= 2.0; rs++; }
  const double m0 = floor(m * 4294967296.0 + 0.5) / 4294967296.0;
  const float scale = (float)(m0 / (double)(1ll << rs));

  const size_t elemsA = (size_t)M * K;
  const size_t elemsB = (size_t)K * N;
  const size_t need = 2 * (elemsA + elemsB) * sizeof(unsigned short);

  dim3 grid((unsigned)(N / BN), (unsigned)(M / BM));
  dim3 block(256);

  if (ws_size >= need) {
    unsigned short* Xh  = (unsigned short*)d_ws;
    unsigned short* Xl  = Xh + elemsA;
    unsigned short* Wth = Xl + elemsA;
    unsigned short* Wtl = Wth + elemsB;

    hipLaunchKernelGGL(convert_split_kernel,
                       dim3((unsigned)(elemsA / 512)), dim3(256), 0, stream,
                       x, Xh, Xl);
    hipLaunchKernelGGL(convert_wT_kernel,
                       dim3((unsigned)(N / 32), (unsigned)(K / 32)),
                       dim3(32, 8), 0, stream, W, Wth, Wtl, (int)K, (int)N);
    hipLaunchKernelGGL(qlinear_gemm_tdm, grid, block, 0, stream,
                       Xh, Xl, Wth, Wtl, b, out,
                       (int)M, (int)N, (int)K, scale);
  } else {
    hipLaunchKernelGGL(qlinear_gemm_fused, grid, block, 0, stream,
                       x, W, b, out, (int)M, (int)N, (int)K, scale);
  }
}
